// PointNet2Regressor_17824114278744
// MI455X (gfx1250) — matrix operand order repaired and checked by reading, not compiled, not run on hardware
//
#include <hip/hip_runtime.h>
#include <hip/hip_bf16.h>
#include <stdint.h>

typedef __attribute__((ext_vector_type(16))) _Float16 v16h;
typedef __attribute__((ext_vector_type(8)))  _Float16 v8h;
typedef __attribute__((ext_vector_type(8)))  float    v8f;
typedef _Float16 h16;

#define BB  2
#define NN  8192
#define NP1 2048
#define NP2 512
#define NP3 128
#define KNB 16

// ---------------------------------------------------------------------------
// xyz prep: (B,3,N) f32 -> (B,N,3) f32 and (B,3,N) f16 (CP copy for fp1 skip)
// ---------------------------------------------------------------------------
__global__ void k_xyz_prep(const float* __restrict__ xbcn, float* __restrict__ xyz,
                           h16* __restrict__ xyz16, int Npts) {
  size_t i = (size_t)blockIdx.x * blockDim.x + threadIdx.x;
  if (i >= (size_t)BB * Npts) return;
  int b = (int)(i / Npts), n = (int)(i % Npts);
  for (int j = 0; j < 3; ++j) {
    float v = xbcn[((size_t)b * 3 + j) * Npts + n];
    xyz[((size_t)b * Npts + n) * 3 + j] = v;
    xyz16[((size_t)b * 3 + j) * Npts + n] = (h16)v;
  }
}

// ---------------------------------------------------------------------------
// Farthest point sampling: one block per batch, dist kept in LDS
// ---------------------------------------------------------------------------
__global__ void k_fps(const float* __restrict__ pts, int Nin, int npoint, int* __restrict__ oidx) {
  int b = blockIdx.x;
  const float* X = pts + (size_t)b * Nin * 3;
  __shared__ float dist[8192];
  __shared__ float rv[256];
  __shared__ int   ri[256];
  __shared__ int   s_far;
  for (int n = threadIdx.x; n < Nin; n += 256) dist[n] = 1e10f;
  if (threadIdx.x == 0) s_far = 0;
  __syncthreads();
  for (int it = 0; it < npoint; ++it) {
    int far = s_far;
    if (threadIdx.x == 0) oidx[b * npoint + it] = far;
    float cx = X[far * 3 + 0], cy = X[far * 3 + 1], cz = X[far * 3 + 2];
    float bv = -1.0f; int bi = 0;
    for (int n = threadIdx.x; n < Nin; n += 256) {
      float dx = X[n * 3 + 0] - cx, dy = X[n * 3 + 1] - cy, dz = X[n * 3 + 2] - cz;
      float d = dx * dx + dy * dy + dz * dz;
      float nd = fminf(dist[n], d);
      dist[n] = nd;
      if (nd > bv) { bv = nd; bi = n; }
    }
    rv[threadIdx.x] = bv; ri[threadIdx.x] = bi;
    __syncthreads();
    for (int st = 128; st > 0; st >>= 1) {
      if (threadIdx.x < st) {
        float ov = rv[threadIdx.x + st]; int oi = ri[threadIdx.x + st];
        if (ov > rv[threadIdx.x] || (ov == rv[threadIdx.x] && oi < ri[threadIdx.x])) {
          rv[threadIdx.x] = ov; ri[threadIdx.x] = oi;
        }
      }
      __syncthreads();
    }
    if (threadIdx.x == 0) s_far = ri[0];
    __syncthreads();
  }
}

__global__ void k_gather3(const float* __restrict__ pts, const int* __restrict__ idx,
                          float* __restrict__ outp, int Nin, int M) {
  size_t i = (size_t)blockIdx.x * blockDim.x + threadIdx.x;
  if (i >= (size_t)BB * M) return;
  int b = (int)(i / M), m = (int)(i % M);
  int n = idx[b * M + m];
  for (int j = 0; j < 3; ++j)
    outp[((size_t)b * M + m) * 3 + j] = pts[((size_t)b * Nin + n) * 3 + j];
}

// ---------------------------------------------------------------------------
// kNN (k=16) of query centers against source cloud (3-D coords)
// ---------------------------------------------------------------------------
__global__ void k_knn16(const float* __restrict__ pts, const float* __restrict__ q,
                        int Nin, int M, int* __restrict__ oidx) {
  size_t i = (size_t)blockIdx.x * blockDim.x + threadIdx.x;
  if (i >= (size_t)BB * M) return;
  int b = (int)(i / M), m = (int)(i % M);
  const float* X = pts + (size_t)b * Nin * 3;
  const float* Q = q + ((size_t)b * M + m) * 3;
  float qx = Q[0], qy = Q[1], qz = Q[2];
  float bd_[16]; int bi_[16];
  for (int j = 0; j < 16; ++j) { bd_[j] = 3.0e38f; bi_[j] = 0; }
  for (int n = 0; n < Nin; ++n) {
    float dx = X[n * 3 + 0] - qx, dy = X[n * 3 + 1] - qy, dz = X[n * 3 + 2] - qz;
    float d2 = dx * dx + dy * dy + dz * dz;
    if (d2 < bd_[15]) {
      int j = 15;
      while (j > 0 && bd_[j - 1] > d2) { bd_[j] = bd_[j - 1]; bi_[j] = bi_[j - 1]; --j; }
      bd_[j] = d2; bi_[j] = n;
    }
  }
  for (int j = 0; j < 16; ++j) oidx[((size_t)b * M + m) * 16 + j] = bi_[j];
}

// ---------------------------------------------------------------------------
// 3-NN + inverse-distance weights for feature propagation
// ---------------------------------------------------------------------------
__global__ void k_knn3(const float* __restrict__ cpts, const float* __restrict__ q,
                       int Mc, int Nf, int* __restrict__ oidx, float* __restrict__ ow) {
  size_t i = (size_t)blockIdx.x * blockDim.x + threadIdx.x;
  if (i >= (size_t)BB * Nf) return;
  int b = (int)(i / Nf), n = (int)(i % Nf);
  const float* X = cpts + (size_t)b * Mc * 3;
  const float* Q = q + ((size_t)b * Nf + n) * 3;
  float qx = Q[0], qy = Q[1], qz = Q[2];
  float bd_[3] = {1e30f, 1e30f, 1e30f}; int bi_[3] = {0, 0, 0};
  for (int m = 0; m < Mc; ++m) {
    float dx = X[m * 3 + 0] - qx, dy = X[m * 3 + 1] - qy, dz = X[m * 3 + 2] - qz;
    float d2 = dx * dx + dy * dy + dz * dz;
    if (d2 < bd_[2]) {
      int j = 2;
      while (j > 0 && bd_[j - 1] > d2) { bd_[j] = bd_[j - 1]; bi_[j] = bi_[j - 1]; --j; }
      bd_[j] = d2; bi_[j] = m;
    }
  }
  float d3[3], w[3]; bool zero = false;
  for (int j = 0; j < 3; ++j) {
    d3[j] = sqrtf(fmaxf(bd_[j], 0.0f) + 1e-24f);
    if (d3[j] <= 1e-12f) zero = true;
  }
  if (zero) { w[0] = 1.0f; w[1] = 0.0f; w[2] = 0.0f; }
  else {
    float s = 0.0f;
    for (int j = 0; j < 3; ++j) { w[j] = 1.0f / (d3[j] + 1e-12f); s += w[j]; }
    for (int j = 0; j < 3; ++j) w[j] /= s;
  }
  for (int j = 0; j < 3; ++j) {
    oidx[((size_t)b * Nf + n) * 3 + j] = bi_[j];
    ow[((size_t)b * Nf + n) * 3 + j] = w[j];
  }
}

// ---------------------------------------------------------------------------
// Build grouped features (per batch): delta-xyz + gathered feature channels,
// channels [3+Cf, Cpad) zero-filled so the GEMM needs no K bounds.
// ---------------------------------------------------------------------------
__global__ void k_group_feat(const float* __restrict__ xyz_b, const float* __restrict__ cents_b,
                             const int* __restrict__ idx_b, const h16* __restrict__ feat_b,
                             h16* __restrict__ gf, int Nin, int M, int Cf, int Cpad) {
  size_t i = (size_t)blockIdx.x * blockDim.x + threadIdx.x;
  size_t P = (size_t)M * 16;
  if (i >= P) return;
  int m = (int)(i >> 4);
  int n = idx_b[i];
  float cx = cents_b[m * 3 + 0], cy = cents_b[m * 3 + 1], cz = cents_b[m * 3 + 2];
  gf[0 * P + i] = (h16)(xyz_b[n * 3 + 0] - cx);
  gf[1 * P + i] = (h16)(xyz_b[n * 3 + 1] - cy);
  gf[2 * P + i] = (h16)(xyz_b[n * 3 + 2] - cz);
  for (int cf = 0; cf < Cf; ++cf)
    gf[(size_t)(3 + cf) * P + i] = feat_b[(size_t)cf * Nin + n];
  for (int c = 3 + Cf; c < Cpad; ++c)
    gf[(size_t)c * P + i] = (h16)0.0f;
}

// ---------------------------------------------------------------------------
// Weight f32 -> f16 with zero padding to [CoP][CiP]; bias f32 padded to CoP
// ---------------------------------------------------------------------------
__global__ void k_wcvt_pad(const float* __restrict__ W, h16* __restrict__ Wh,
                           int Co, int Ci, int CoP, int CiP) {
  size_t i = (size_t)blockIdx.x * blockDim.x + threadIdx.x;
  if (i >= (size_t)CoP * CiP) return;
  int row = (int)(i / CiP), col = (int)(i % CiP);
  float v = (row < Co && col < Ci) ? W[(size_t)row * Ci + col] : 0.0f;
  Wh[i] = (h16)v;
}

__global__ void k_bias_pad(const float* __restrict__ bsrc, float* __restrict__ bdst,
                           int Co, int CoP) {
  int i = blockIdx.x * blockDim.x + threadIdx.x;
  if (i >= CoP) return;
  bdst[i] = (i < Co) ? bsrc[i] : 0.0f;
}

// ---------------------------------------------------------------------------
// WMMA GEMM:  Y[CoP][P] = Wh[CoP][Cip] * X[Cip][P] + biasPad   (f16, f32 acc)
// Block = 8 waves = one 16(Co) x 128(P) tile; LDS-staged, fully branch-free
// (weights, bias, X and Y all padded). Requires P%128==0, Cip%32==0, CoP%16==0.
// ---------------------------------------------------------------------------
__global__ void k_gemm(const h16* __restrict__ Wh, const float* __restrict__ biasPad,
                       const h16* __restrict__ X, h16* __restrict__ Y,
                       int Cip, int P) {
  __shared__ h16 sA[16][32];    // [co_row][k]
  __shared__ h16 sB[128][32];   // [p_local][k]  (transposed so lane reads are contiguous)
  int t = threadIdx.x;
  int wave = t >> 5, lane = t & 31;
  int m = lane & 15, half = lane >> 4;
  int co0 = blockIdx.y * 16;
  int p0 = blockIdx.x * 128;
  int pl = wave * 16 + m;       // this lane's B column within the block tile

  // A staging coords (2 f16 = one b32 per thread)
  int arow = (t * 2) >> 5;
  int akk  = (t * 2) & 31;

  v8f acc;
#pragma unroll
  for (int v = 0; v < 8; ++v) acc[v] = 0.0f;

  for (int k0 = 0; k0 < Cip; k0 += 32) {
    __syncthreads();
    // ---- stage A tile (16x32): one unconditional b32 per thread
    *(unsigned*)&sA[arow][akk] =
        *(const unsigned*)(Wh + (size_t)(co0 + arow) * Cip + k0 + akk);
    // ---- stage B tile (32k x 128p): unconditional b128 loads, transposed store
#pragma unroll
    for (int r = 0; r < 2; ++r) {
      int c = t + 256 * r;      // 0..511 chunk id
      int kk = c >> 4;          // 0..31
      int pc = (c & 15) * 8;    // 0..120
      v8h v = *(const v8h*)(X + (size_t)(k0 + kk) * P + p0 + pc);
#pragma unroll
      for (int j = 0; j < 8; ++j) sB[pc + j][kk] = v[j];
    }
    __syncthreads();
    // ---- build fragments (ISA VGPR layouts) and issue WMMA
    v16h a, bf;
#pragma unroll
    for (int jj = 0; jj < 8; ++jj) {
      int kb = ((jj < 4) ? 0 : 16) + half * 8 + (jj & 3) * 2;
      a[2 * jj]     = sA[m][kb];
      a[2 * jj + 1] = sA[m][kb + 1];
    }
    {
      const v8h* pB = (const v8h*)&sB[pl][0];
      v8h b0 = pB[half], b1 = pB[2 + half];
#pragma unroll
      for (int j = 0; j < 8; ++j) { bf[j] = b0[j]; bf[8 + j] = b1[j]; }
    }
    acc = __builtin_amdgcn_wmma_f32_16x16x32_f16(false, a, false, bf,
                                                 (short)0, acc, false, false);
  }

  int n = lane & 15;
#pragma unroll
  for (int v = 0; v < 8; ++v) {
    int co = co0 + v + 8 * half;
    int p = p0 + wave * 16 + n;
    Y[(size_t)co * P + p] = (h16)(acc[v] + biasPad[co]);
  }
}

// ---------------------------------------------------------------------------
// GroupNorm (g=32): per-batch stats over (C/32 channels x all positions)
// ---------------------------------------------------------------------------
__global__ void k_gn_stats(const h16* __restrict__ X, float* __restrict__ stats,
                           int C, size_t P) {
  int grp = blockIdx.x;
  int cpg = C / 32;
  const h16* base = X + (size_t)grp * cpg * P;
  size_t n = (size_t)cpg * P;
  float s = 0.0f, q = 0.0f;
  for (size_t i = threadIdx.x; i < n; i += blockDim.x) {
    float v = (float)base[i];
    s += v; q += v * v;
  }
  __shared__ float ss[256], sq[256];
  ss[threadIdx.x] = s; sq[threadIdx.x] = q;
  __syncthreads();
  for (int st = 128; st > 0; st >>= 1) {
    if (threadIdx.x < st) { ss[threadIdx.x] += ss[threadIdx.x + st]; sq[threadIdx.x] += sq[threadIdx.x + st]; }
    __syncthreads();
  }
  if (threadIdx.x == 0) {
    float mean = ss[0] / (float)n;
    float var = fmaxf(sq[0] / (float)n - mean * mean, 0.0f);
    stats[grp * 2 + 0] = mean;
    stats[grp * 2 + 1] = rsqrtf(var + 1e-5f);
  }
}

__global__ void k_gn_apply(h16* __restrict__ X, const float* __restrict__ stats,
                           const float* __restrict__ g, const float* __restrict__ be,
                           int C, size_t P, int relu) {
  size_t i = (size_t)blockIdx.x * blockDim.x + threadIdx.x;
  size_t tot = (size_t)C * P;
  if (i >= tot) return;
  int c = (int)(i / P);
  int grp = c / (C / 32);
  float mean = stats[grp * 2 + 0], rstd = stats[grp * 2 + 1];
  float v = ((float)X[i] - mean) * rstd * g[c] + be[c];
  if (relu) v = fmaxf(v, 0.0f);
  X[i] = (h16)v;
}

// ---------------------------------------------------------------------------
// Max over K=16 neighbor dim (optional residual add)
// ---------------------------------------------------------------------------
__global__ void k_maxpool(const h16* __restrict__ X, const h16* __restrict__ resid,
                          h16* __restrict__ Y, int C, int M) {
  size_t i = (size_t)blockIdx.x * blockDim.x + threadIdx.x;
  if (i >= (size_t)C * M) return;
  int c = (int)(i / M), m = (int)(i % M);
  const h16* p = X + (size_t)c * M * 16 + (size_t)m * 16;
  float mx = -1e30f;
  for (int k = 0; k < 16; ++k) mx = fmaxf(mx, (float)p[k]);
  if (resid) mx += (float)resid[(size_t)c * M + m];
  Y[(size_t)c * M + m] = (h16)mx;
}

// ---------------------------------------------------------------------------
// FP interpolation + skip concat (per batch): out[Cpad][Nf], pad rows zero
// ---------------------------------------------------------------------------
__global__ void k_interp(const h16* __restrict__ featC, const int* __restrict__ idx3,
                         const float* __restrict__ w3, const h16* __restrict__ skip,
                         h16* __restrict__ out, int Cc, int Mc, int Cs, int Nf, int Cpad) {
  size_t i = (size_t)blockIdx.x * blockDim.x + threadIdx.x;
  if (i >= (size_t)Cpad * Nf) return;
  int c = (int)(i / Nf), n = (int)(i % Nf);
  float v;
  if (c < Cc) {
    const int* I = idx3 + (size_t)n * 3;
    const float* W = w3 + (size_t)n * 3;
    v = (float)featC[(size_t)c * Mc + I[0]] * W[0]
      + (float)featC[(size_t)c * Mc + I[1]] * W[1]
      + (float)featC[(size_t)c * Mc + I[2]] * W[2];
  } else if (c < Cc + Cs) {
    v = (float)skip[(size_t)(c - Cc) * Nf + n];
  } else {
    v = 0.0f;
  }
  out[(size_t)c * Nf + n] = (h16)v;
}

// ---------------------------------------------------------------------------
// EdgeConv helpers (per batch)
// ---------------------------------------------------------------------------
__global__ void k_xx(const h16* __restrict__ X, float* __restrict__ xx, int C, int Npts) {
  size_t n = (size_t)blockIdx.x * blockDim.x + threadIdx.x;
  if (n >= (size_t)Npts) return;
  float s = 0.0f;
  for (int c = 0; c < C; ++c) {
    float v = (float)X[(size_t)c * Npts + n];
    s += v * v;
  }
  xx[n] = s;
}

__global__ void k_knn_full(const h16* __restrict__ X, const float* __restrict__ xx,
                           int* __restrict__ oidx, int C, int Npts) {
  __shared__ h16 sxi[128][128];
  int n = blockIdx.x * blockDim.x + threadIdx.x;
  int t = threadIdx.x;
  if (n < Npts)
    for (int c = 0; c < C; ++c) sxi[t][c] = X[(size_t)c * Npts + n];
  if (n >= Npts) return;
  float xn = xx[n];
  float bd_[16]; int bi_[16];
  for (int j = 0; j < 16; ++j) { bd_[j] = 3.0e38f; bi_[j] = 0; }
  for (int m = 0; m < Npts; ++m) {
    float dot = 0.0f;
    for (int c = 0; c < C; ++c)
      dot += (float)sxi[t][c] * (float)X[(size_t)c * Npts + m];
    float d = fmaxf(xn + xx[m] - 2.0f * dot, 0.0f);
    if (d < bd_[15]) {
      int j = 15;
      while (j > 0 && bd_[j - 1] > d) { bd_[j] = bd_[j - 1]; bi_[j] = bi_[j - 1]; --j; }
      bd_[j] = d; bi_[j] = m;
    }
  }
  for (int j = 0; j < 16; ++j) oidx[(size_t)n * 16 + j] = bi_[j];
}

__global__ void k_edge(const h16* __restrict__ X, const int* __restrict__ idx,
                       h16* __restrict__ E, int C, int Npts) {
  size_t i = (size_t)blockIdx.x * blockDim.x + threadIdx.x;
  size_t P = (size_t)Npts * 16;
  if (i >= P) return;
  int n = (int)(i >> 4);
  int nj = idx[i];
  for (int c = 0; c < C; ++c) {
    float xi = (float)X[(size_t)c * Npts + n];
    float xj = (float)X[(size_t)c * Npts + nj];
    E[(size_t)c * P + i] = (h16)xi;
    E[(size_t)(C + c) * P + i] = (h16)(xj - xi);
  }
}

__global__ void k_add(h16* __restrict__ A, const h16* __restrict__ Bx, size_t n) {
  size_t i = (size_t)blockIdx.x * blockDim.x + threadIdx.x;
  if (i >= n) return;
  A[i] = (h16)((float)A[i] + (float)Bx[i]);
}

// pb/ub are channel-padded to 16 rows per batch
__global__ void k_out(const h16* __restrict__ pb, const h16* __restrict__ ub,
                      float* __restrict__ out, int Npts) {
  size_t i = (size_t)blockIdx.x * blockDim.x + threadIdx.x;
  if (i >= (size_t)BB * Npts) return;
  int b = (int)(i / Npts), n = (int)(i % Npts);
  out[((size_t)b * 4 + 0) * Npts + n] = (float)pb[(size_t)b * 16 * Npts + n];
  for (int j = 0; j < 3; ++j)
    out[((size_t)b * 4 + 1 + j) * Npts + n] = (float)ub[((size_t)b * 16 + j) * Npts + n];
}

// ===========================================================================
// Host orchestration
// ===========================================================================
struct M2 { const float *w1, *b1, *g1, *be1, *w2, *b2, *g2, *be2; };
struct HD { M2 m; const float *w3, *b3; };

extern "C" void kernel_launch(void* const* d_in, const int* in_sizes, int n_in,
                              void* d_out, int out_size, void* d_ws, size_t ws_size,
                              hipStream_t stream) {
  (void)out_size; (void)ws_size;
  // ---- locate xyz (insertion order => first; fully sorted flatten => last)
  int xyzI = (in_sizes[0] == BB * 3 * NN) ? 0 : (n_in - 1);
  const float* xbcn = (const float*)d_in[xyzI];
  const float* prm[128]; int psz[128]; int npar = 0;
  for (int i = 0; i < n_in && npar < 128; ++i)
    if (i != xyzI) { prm[npar] = (const float*)d_in[i]; psz[npar] = in_sizes[i]; ++npar; }
  bool insertion = (psz[0] == 64 * 3);   // sa1.w1 first under insertion order

  auto m2_ins = [&](int o) { M2 p; p.w1 = prm[o]; p.b1 = prm[o + 1]; p.g1 = prm[o + 2]; p.be1 = prm[o + 3];
                             p.w2 = prm[o + 4]; p.b2 = prm[o + 5]; p.g2 = prm[o + 6]; p.be2 = prm[o + 7]; return p; };
  auto m2_srt = [&](int o) { M2 p; p.b1 = prm[o]; p.b2 = prm[o + 1]; p.be1 = prm[o + 2]; p.be2 = prm[o + 3];
                             p.g1 = prm[o + 4]; p.g2 = prm[o + 5]; p.w1 = prm[o + 6]; p.w2 = prm[o + 7]; return p; };
  auto hd_ins = [&](int o) { HD h; h.m = m2_ins(o); h.w3 = prm[o + 8]; h.b3 = prm[o + 9]; return h; };
  auto hd_srt = [&](int o) { HD h; h.m.b1 = prm[o]; h.m.b2 = prm[o + 1]; h.b3 = prm[o + 2];
                             h.m.be1 = prm[o + 3]; h.m.be2 = prm[o + 4]; h.m.g1 = prm[o + 5]; h.m.g2 = prm[o + 6];
                             h.m.w1 = prm[o + 7]; h.m.w2 = prm[o + 8]; h.w3 = prm[o + 9]; return h; };

  M2 sa1, sa2, sa3, fp3, fp2, fp1, ref0, ref1; HD ph, uh;
  const float *fz_w, *fz_b, *fz_g, *fz_be;
  if (insertion) {
    sa1 = m2_ins(0);  sa2 = m2_ins(8);  sa3 = m2_ins(16);
    fp3 = m2_ins(24); fp2 = m2_ins(32); fp1 = m2_ins(40);
    ref0 = m2_ins(48); ref1 = m2_ins(56);
    fz_w = prm[64]; fz_b = prm[65]; fz_g = prm[66]; fz_be = prm[67];
    ph = hd_ins(68); uh = hd_ins(78);
  } else {
    fp1 = m2_srt(0); fp2 = m2_srt(8); fp3 = m2_srt(16);
    fz_b = prm[24]; fz_be = prm[25]; fz_g = prm[26]; fz_w = prm[27];
    ph = hd_srt(28);
    ref0 = m2_srt(38); ref1 = m2_srt(46);
    sa1 = m2_srt(54); sa2 = m2_srt(62); sa3 = m2_srt(70);
    uh = hd_srt(78);
  }

  // ---- workspace bump allocator
  char* wsp = (char*)d_ws; size_t off = 0;
  auto alloc = [&](size_t bytes) -> void* { void* p = wsp + off; off = (off + bytes + 255) & ~(size_t)255; return p; };
  float* xyz   = (float*)alloc((size_t)BB * NN * 3 * 4);
  h16*   xyz16 = (h16*)alloc((size_t)BB * 3 * NN * 2);
  float* l1x = (float*)alloc((size_t)BB * NP1 * 3 * 4);
  float* l2x = (float*)alloc((size_t)BB * NP2 * 3 * 4);
  float* l3x = (float*)alloc((size_t)BB * NP3 * 3 * 4);
  int* fidx  = (int*)alloc((size_t)BB * NP1 * 4);
  int* kidx  = (int*)alloc((size_t)BB * NP1 * 16 * 4);
  int* kidxF = (int*)alloc((size_t)BB * NN * 16 * 4);
  int* idx3  = (int*)alloc((size_t)BB * NN * 3 * 4);
  float* w3b = (float*)alloc((size_t)BB * NN * 3 * 4);
  h16* l1f = (h16*)alloc((size_t)BB * 128 * NP1 * 2);
  h16* l2f = (h16*)alloc((size_t)BB * 256 * NP2 * 2);
  h16* l3f = (h16*)alloc((size_t)BB * 512 * NP3 * 2);
  h16* f2  = (h16*)alloc((size_t)BB * 256 * NP2 * 2);
  h16* f1  = (h16*)alloc((size_t)BB * 128 * NP1 * 2);
  h16* fbuf = (h16*)alloc((size_t)BB * 128 * NN * 2);
  h16* g1b  = (h16*)alloc((size_t)BB * 128 * NN * 2);
  h16* g2b  = (h16*)alloc((size_t)BB * 128 * NN * 2);
  h16* gz   = (h16*)alloc((size_t)BB * 128 * NN * 2);
  float* xxb = (float*)alloc((size_t)BB * NN * 4);
  float* statsB = (float*)alloc(32 * 2 * 4);
  h16* wbuf = (h16*)alloc((size_t)256 * 768 * 2);       // largest padded weight (fp3.w1)
  float* bbuf = (float*)alloc((size_t)512 * 4);         // padded bias
  h16* h1 = (h16*)alloc((size_t)BB * 128 * NN * 2);
  h16* h2 = (h16*)alloc((size_t)BB * 64 * NN * 2);
  h16* pb = (h16*)alloc((size_t)BB * 16 * NN * 2);      // head out, 16 padded rows
  h16* ub = (h16*)alloc((size_t)BB * 16 * NN * 2);
  h16* aIn  = (h16*)alloc((size_t)288 * NN * 16 * 2);   // largest padded phase input
  h16* aMid = (h16*)alloc((size_t)128 * NN * 16 * 2);
  h16* aOut = (h16*)alloc((size_t)128 * NN * 16 * 2);

  auto cd = [](size_t n, int blk) { return (unsigned)((n + blk - 1) / blk); };

  // Ci = logical input channels of the weight; Cip = padded (%32) channel rows in X.
  // Y gets CoP (=ceil16(Co)) rows; pad rows receive zeros (padded W + zero bias).
  auto gemm = [&](const float* Wf, const float* bias, const h16* X, h16* Y,
                  int Co, int Ci, int Cip, size_t P) {
    int CoP = (Co + 15) & ~15;
    k_wcvt_pad<<<cd((size_t)CoP * Cip, 256), 256, 0, stream>>>(Wf, wbuf, Co, Ci, CoP, Cip);
    k_bias_pad<<<cd((size_t)CoP, 64), 64, 0, stream>>>(bias, bbuf, Co, CoP);
    dim3 g((unsigned)(P / 128), (unsigned)(CoP / 16));
    k_gemm<<<g, 256, 0, stream>>>(wbuf, bbuf, X, Y, Cip, (int)P);
  };
  auto gn = [&](h16* X, const float* gam, const float* bet, int C, size_t P, int relu) {
    k_gn_stats<<<32, 256, 0, stream>>>(X, statsB, C, P);
    k_gn_apply<<<cd((size_t)C * P, 256), 256, 0, stream>>>(X, statsB, gam, bet, C, P, relu);
  };
  auto mlp2 = [&](const M2& p, const h16* X, h16* mid, h16* out,
                  int Ci, int Cip, int Cm, int Co, size_t P) {
    gemm(p.w1, p.b1, X, mid, Cm, Ci, Cip, P);
    gn(mid, p.g1, p.be1, Cm, P, 1);
    gemm(p.w2, p.b2, mid, out, Co, Cm, Cm, P);
    gn(out, p.g2, p.be2, Co, P, 1);
  };

  // ---- prep
  k_xyz_prep<<<cd((size_t)BB * NN, 256), 256, 0, stream>>>(xbcn, xyz, xyz16, NN);

  // ---- SA1: 8192 -> 2048 centers, cin=3(pad32) -> 64 -> 128
  k_fps<<<BB, 256, 0, stream>>>(xyz, NN, NP1, fidx);
  k_gather3<<<cd((size_t)BB * NP1, 256), 256, 0, stream>>>(xyz, fidx, l1x, NN, NP1);
  k_knn16<<<cd((size_t)BB * NP1, 128), 128, 0, stream>>>(xyz, l1x, NN, NP1, kidx);
  for (int b = 0; b < BB; ++b) {
    size_t P = (size_t)NP1 * 16;
    k_group_feat<<<cd(P, 256), 256, 0, stream>>>(xyz + (size_t)b * NN * 3, l1x + (size_t)b * NP1 * 3,
                                                 kidx + (size_t)b * NP1 * 16, (const h16*)nullptr, aIn,
                                                 NN, NP1, 0, 32);
    mlp2(sa1, aIn, aMid, aOut, 3, 32, 64, 128, P);
    k_maxpool<<<cd((size_t)128 * NP1, 256), 256, 0, stream>>>(aOut, (const h16*)nullptr,
                                                              l1f + (size_t)b * 128 * NP1, 128, NP1);
  }

  // ---- SA2: 2048 -> 512, cin=131(pad160) -> 128 -> 256
  k_fps<<<BB, 256, 0, stream>>>(l1x, NP1, NP2, fidx);
  k_gather3<<<cd((size_t)BB * NP2, 256), 256, 0, stream>>>(l1x, fidx, l2x, NP1, NP2);
  k_knn16<<<cd((size_t)BB * NP2, 128), 128, 0, stream>>>(l1x, l2x, NP1, NP2, kidx);
  for (int b = 0; b < BB; ++b) {
    size_t P = (size_t)NP2 * 16;
    k_group_feat<<<cd(P, 256), 256, 0, stream>>>(l1x + (size_t)b * NP1 * 3, l2x + (size_t)b * NP2 * 3,
                                                 kidx + (size_t)b * NP2 * 16, l1f + (size_t)b * 128 * NP1,
                                                 aIn, NP1, NP2, 128, 160);
    mlp2(sa2, aIn, aMid, aOut, 131, 160, 128, 256, P);
    k_maxpool<<<cd((size_t)256 * NP2, 256), 256, 0, stream>>>(aOut, (const h16*)nullptr,
                                                              l2f + (size_t)b * 256 * NP2, 256, NP2);
  }

  // ---- SA3: 512 -> 128, cin=259(pad288) -> 256 -> 512
  k_fps<<<BB, 256, 0, stream>>>(l2x, NP2, NP3, fidx);
  k_gather3<<<cd((size_t)BB * NP3, 256), 256, 0, stream>>>(l2x, fidx, l3x, NP2, NP3);
  k_knn16<<<cd((size_t)BB * NP3, 128), 128, 0, stream>>>(l2x, l3x, NP2, NP3, kidx);
  for (int b = 0; b < BB; ++b) {
    size_t P = (size_t)NP3 * 16;
    k_group_feat<<<cd(P, 256), 256, 0, stream>>>(l2x + (size_t)b * NP2 * 3, l3x + (size_t)b * NP3 * 3,
                                                 kidx + (size_t)b * NP3 * 16, l2f + (size_t)b * 256 * NP2,
                                                 aIn, NP2, NP3, 256, 288);
    mlp2(sa3, aIn, aMid, aOut, 259, 288, 256, 512, P);
    k_maxpool<<<cd((size_t)512 * NP3, 256), 256, 0, stream>>>(aOut, (const h16*)nullptr,
                                                              l3f + (size_t)b * 512 * NP3, 512, NP3);
  }

  // ---- FP3: interp l3f -> l2x, concat l2f (768 -> 256 -> 256) => f2
  k_knn3<<<cd((size_t)BB * NP2, 128), 128, 0, stream>>>(l3x, l2x, NP3, NP2, idx3, w3b);
  for (int b = 0; b < BB; ++b) {
    k_interp<<<cd((size_t)768 * NP2, 256), 256, 0, stream>>>(l3f + (size_t)b * 512 * NP3,
        idx3 + (size_t)b * NP2 * 3, w3b + (size_t)b * NP2 * 3, l2f + (size_t)b * 256 * NP2,
        aIn, 512, NP3, 256, NP2, 768);
    mlp2(fp3, aIn, aMid, f2 + (size_t)b * 256 * NP2, 768, 768, 256, 256, (size_t)NP2);
  }

  // ---- FP2: interp f2 -> l1x, concat l1f (384 -> 128 -> 128) => f1
  k_knn3<<<cd((size_t)BB * NP1, 128), 128, 0, stream>>>(l2x, l1x, NP2, NP1, idx3, w3b);
  for (int b = 0; b < BB; ++b) {
    k_interp<<<cd((size_t)384 * NP1, 256), 256, 0, stream>>>(f2 + (size_t)b * 256 * NP2,
        idx3 + (size_t)b * NP1 * 3, w3b + (size_t)b * NP1 * 3, l1f + (size_t)b * 128 * NP1,
        aIn, 256, NP2, 128, NP1, 384);
    mlp2(fp2, aIn, aMid, f1 + (size_t)b * 128 * NP1, 384, 384, 128, 128, (size_t)NP1);
  }

  // ---- FP1: interp f1 -> xyz, concat xyz (131(pad160) -> 128 -> 128) => f
  k_knn3<<<cd((size_t)BB * NN, 128), 128, 0, stream>>>(l1x, xyz, NP1, NN, idx3, w3b);
  for (int b = 0; b < BB; ++b) {
    k_interp<<<cd((size_t)160 * NN, 256), 256, 0, stream>>>(f1 + (size_t)b * 128 * NP1,
        idx3 + (size_t)b * NN * 3, w3b + (size_t)b * NN * 3, xyz16 + (size_t)b * 3 * NN,
        aIn, 128, NP1, 3, NN, 160);
    mlp2(fp1, aIn, aMid, fbuf + (size_t)b * 128 * NN, 131, 160, 128, 128, (size_t)NN);
  }

  // ---- EdgeConvs (channel-space kNN over full N)
  auto edgeconv = [&](const M2& p, const h16* Xin, h16* Yout) {
    for (int b = 0; b < BB; ++b) {
      const h16* Xb = Xin + (size_t)b * 128 * NN;
      k_xx<<<cd((size_t)NN, 256), 256, 0, stream>>>(Xb, xxb + (size_t)b * NN, 128, NN);
      k_knn_full<<<cd((size_t)NN, 128), 128, 0, stream>>>(Xb, xxb + (size_t)b * NN,
                                                          kidxF + (size_t)b * NN * 16, 128, NN);
      k_edge<<<cd((size_t)NN * 16, 256), 256, 0, stream>>>(Xb, kidxF + (size_t)b * NN * 16, aIn, 128, NN);
      size_t P = (size_t)NN * 16;
      mlp2(p, aIn, aMid, aOut, 256, 256, 128, 128, P);
      k_maxpool<<<cd((size_t)128 * NN, 256), 256, 0, stream>>>(aOut, Xb, Yout + (size_t)b * 128 * NN, 128, NN);
    }
  };
  edgeconv(ref0, fbuf, g1b);
  edgeconv(ref1, g1b, g2b);

  // ---- fuse: relu(gn(conv(g)));  f = f + g
  for (int b = 0; b < BB; ++b) {
    gemm(fz_w, fz_b, g2b + (size_t)b * 128 * NN, gz + (size_t)b * 128 * NN, 128, 128, 128, (size_t)NN);
    gn(gz + (size_t)b * 128 * NN, fz_g, fz_be, 128, (size_t)NN, 1);
  }
  k_add<<<cd((size_t)BB * 128 * NN, 256), 256, 0, stream>>>(fbuf, gz, (size_t)BB * 128 * NN);

  // ---- heads (final conv writes 16 padded channel rows)
  auto head = [&](const HD& h, h16* outBuf, int Co3) {
    for (int b = 0; b < BB; ++b) {
      const h16* fb = fbuf + (size_t)b * 128 * NN;
      h16* h1b = h1 + (size_t)b * 128 * NN;
      h16* h2b = h2 + (size_t)b * 64 * NN;
      gemm(h.m.w1, h.m.b1, fb, h1b, 128, 128, 128, (size_t)NN);
      gn(h1b, h.m.g1, h.m.be1, 128, (size_t)NN, 1);
      gemm(h.m.w2, h.m.b2, h1b, h2b, 64, 128, 128, (size_t)NN);
      gn(h2b, h.m.g2, h.m.be2, 64, (size_t)NN, 1);
      gemm(h.w3, h.b3, h2b, outBuf + (size_t)b * 16 * NN, Co3, 64, 64, (size_t)NN);
    }
  };
  head(ph, pb, 1);
  head(uh, ub, 3);

  k_out<<<cd((size_t)BB * NN, 256), 256, 0, stream>>>(pb, ub, (float*)d_out, NN);
}
